// UDecoder_43671227466120
// MI455X (gfx1250) — matrix operand order repaired and checked by reading, not compiled, not run on hardware
//
#include <hip/hip_runtime.h>

// ---------------- types ----------------
typedef __bf16 bf16_t;
typedef __attribute__((ext_vector_type(16))) __bf16        v16bf;
typedef __attribute__((ext_vector_type(8)))  float         v8f;
typedef __attribute__((ext_vector_type(4)))  unsigned int  u32x4;
typedef __attribute__((ext_vector_type(4)))  float         f32x4;

union FragBf { v16bf v; u32x4 u[2]; };

#define N_NODES 8192
#define M_NODES 8192
#define D_MODEL 512
#define N_HEADS 8
#define D_FF    2048

// ---------------- elementwise helpers ----------------
__global__ __launch_bounds__(256) void fill_zero_kernel(float* __restrict__ p, int total) {
    int i = blockIdx.x * 256 + threadIdx.x;
    if (i < total) p[i] = 0.0f;
}

__global__ __launch_bounds__(256) void f2bf_kernel(const float* __restrict__ a,
                                                   bf16_t* __restrict__ o, int total) {
    int i = blockIdx.x * 256 + threadIdx.x;
    if (i < total) o[i] = (bf16_t)a[i];
}

// Wt[n*K + k] = (bf16) W[k*Nn + n]
__global__ __launch_bounds__(256) void transpose_bf16_kernel(const float* __restrict__ W,
                                                             bf16_t* __restrict__ Wt,
                                                             int K, int Nn, int total) {
    int i = blockIdx.x * 256 + threadIdx.x;
    if (i >= total) return;
    int n = i / K;
    int k = i - n * K;
    Wt[i] = (bf16_t)W[(size_t)k * Nn + n];
}

// ---------------- LayerNorm (one block per row) -> bf16 ----------------
__global__ __launch_bounds__(256) void ln_bf16_kernel(const float* __restrict__ x,
                                                      const float* __restrict__ g,
                                                      const float* __restrict__ b,
                                                      bf16_t* __restrict__ out, int D) {
    int row = blockIdx.x;
    const float* xr = x + (size_t)row * D;
    __shared__ float red0[8], red1[8];

    float s = 0.f, s2 = 0.f;
    for (int i = threadIdx.x; i < D; i += 256) { float t = xr[i]; s += t; s2 += t * t; }
    for (int off = 16; off > 0; off >>= 1) {
        s  += __shfl_xor(s,  off, 32);
        s2 += __shfl_xor(s2, off, 32);
    }
    int wave = threadIdx.x >> 5;
    if ((threadIdx.x & 31) == 0) { red0[wave] = s; red1[wave] = s2; }
    __syncthreads();
    if (threadIdx.x < 32) {
        float a  = (threadIdx.x < 8) ? red0[threadIdx.x] : 0.f;
        float a2 = (threadIdx.x < 8) ? red1[threadIdx.x] : 0.f;
        for (int off = 4; off > 0; off >>= 1) {
            a  += __shfl_xor(a,  off, 32);
            a2 += __shfl_xor(a2, off, 32);
        }
        if (threadIdx.x == 0) { red0[0] = a; red1[0] = a2; }
    }
    __syncthreads();
    float mu  = red0[0] / (float)D;
    float var = red1[0] / (float)D - mu * mu;
    float rs  = rsqrtf(var + 1e-6f);
    for (int i = threadIdx.x; i < D; i += 256) {
        float t = (xr[i] - mu) * rs * g[i] + b[i];
        out[(size_t)row * D + i] = (bf16_t)t;
    }
}

// ---------------- WMMA bf16 GEMM ----------------
// C[M x Nout] = A[M x K](bf16, row major) * Bt[Nout x K](bf16, row major, pre-transposed)
// epilogue: + bias[n], optional relu, optional += addsrc, write f32 and/or bf16.
// Block tile 128x128, 8 waves, wave tile 32x64, K step 32.
#define LDT 48   // padded LDS row stride (elements); 96B rows, 16B aligned

__global__ __launch_bounds__(256) void gemm_bf16_kernel(const bf16_t* __restrict__ A,
                                                        const bf16_t* __restrict__ Bt,
                                                        const float*  __restrict__ bias,
                                                        const float*  __restrict__ addsrc,
                                                        float*        __restrict__ outf,
                                                        bf16_t*       __restrict__ outbf,
                                                        int K, int Nout, int relu) {
    __shared__ bf16_t As[128 * LDT];
    __shared__ bf16_t Bs[128 * LDT];

    const int tid  = threadIdx.x;
    const int lane = tid & 31;
    const int wave = tid >> 5;
    const int wm   = wave >> 1;           // 0..3 -> M offset 32*wm
    const int wn   = wave & 1;            // 0..1 -> N offset 64*wn
    const int gm0  = blockIdx.y * 128;
    const int gn0  = blockIdx.x * 128;

    const int lr = lane & 15;             // row (A) / col (B) within fragment
    const int lh = lane >> 4;             // lane half
    const int ka = lh * 8;                // A: K pairs at ka.. and ka+16..

    v8f c[2][4];
    #pragma unroll
    for (int i = 0; i < 2; ++i)
        #pragma unroll
        for (int j = 0; j < 4; ++j)
            c[i][j] = (v8f){0.f,0.f,0.f,0.f,0.f,0.f,0.f,0.f};

    for (int k0 = 0; k0 < K; k0 += 32) {
        // stage 128x32 A tile and 128x32 Bt tile (16B per thread-chunk, 512 chunks each)
        #pragma unroll
        for (int it = 0; it < 2; ++it) {
            int ch  = tid + it * 256;     // 0..511
            int row = ch >> 2;
            int sub = ch & 3;
            *(u32x4*)(&As[row * LDT + sub * 8]) =
                *(const u32x4*)(A  + (size_t)(gm0 + row) * K + k0 + sub * 8);
            *(u32x4*)(&Bs[row * LDT + sub * 8]) =
                *(const u32x4*)(Bt + (size_t)(gn0 + row) * K + k0 + sub * 8);
        }
        __syncthreads();

        FragBf a[2], b[4];
        // A fragment: lane lr = row, K = {ka..ka+7} and {ka+16..ka+23} (ISA 16-bit A layout)
        #pragma unroll
        for (int i = 0; i < 2; ++i) {
            const bf16_t* p = &As[(wm * 32 + i * 16 + lr) * LDT];
            a[i].u[0] = *(const u32x4*)(p + ka);
            a[i].u[1] = *(const u32x4*)(p + ka + 16);
        }
        // B fragment: lane lr = column, K = {ka..ka+7} and {ka+16..ka+23} (same order as A)
        #pragma unroll
        for (int j = 0; j < 4; ++j) {
            const bf16_t* p = &Bs[(wn * 64 + j * 16 + lr) * LDT + ka];
            b[j].u[0] = *(const u32x4*)(p);
            b[j].u[1] = *(const u32x4*)(p + 16);
        }

        #pragma unroll
        for (int i = 0; i < 2; ++i)
            #pragma unroll
            for (int j = 0; j < 4; ++j)
                c[i][j] = __builtin_amdgcn_wmma_f32_16x16x32_bf16(
                    false, a[i].v, false, b[j].v, (short)0, c[i][j], false, false);
        __syncthreads();
    }

    // epilogue
    #pragma unroll
    for (int i = 0; i < 2; ++i) {
        #pragma unroll
        for (int j = 0; j < 4; ++j) {
            int n = gn0 + wn * 64 + j * 16 + lr;
            float bv = bias ? bias[n] : 0.f;
            #pragma unroll
            for (int vg = 0; vg < 8; ++vg) {
                int m = gm0 + wm * 32 + i * 16 + lh * 8 + vg;
                float val = c[i][j][vg] + bv;
                if (relu) val = val > 0.f ? val : 0.f;
                size_t idx = (size_t)m * Nout + n;
                if (addsrc) val += addsrc[idx];
                if (outf)  outf[idx]  = val;
                if (outbf) outbf[idx] = (bf16_t)val;
            }
        }
    }
}

// ---------------- edge attention (one wave per edge) ----------------
// score[h] = exp(clip(k[src,h,:].q[dst,h,:]/8, -10, 10)); att[dst] += score*v[src]; z[dst,h] += score
__global__ __launch_bounds__(256) void edge_attn_kernel(const float* __restrict__ q,
                                                        const float* __restrict__ k,
                                                        const float* __restrict__ v,
                                                        const int*   __restrict__ src,
                                                        const int*   __restrict__ dst,
                                                        float* __restrict__ att,
                                                        float* __restrict__ z, int E) {
    int e = blockIdx.x * 8 + (threadIdx.x >> 5);
    if (e >= E) return;
    int lane = threadIdx.x & 31;          // 16 elems per lane; 4 lanes per head
    int s_ = src[e];
    int d_ = dst[e];

    const f32x4* kp = (const f32x4*)(k + (size_t)s_ * D_MODEL + lane * 16);
    const f32x4* qp = (const f32x4*)(q + (size_t)d_ * D_MODEL + lane * 16);
    float acc = 0.f;
    #pragma unroll
    for (int t = 0; t < 4; ++t) {
        f32x4 a = kp[t], b = qp[t];
        acc += a.x * b.x + a.y * b.y + a.z * b.z + a.w * b.w;
    }
    acc += __shfl_xor(acc, 1, 32);        // reduce across the 4 lanes of this head
    acc += __shfl_xor(acc, 2, 32);
    float sc = acc * 0.125f;              // 1/sqrt(64)
    sc = fminf(10.f, fmaxf(-10.f, sc));
    sc = __expf(sc);

    const f32x4* vp = (const f32x4*)(v + (size_t)s_ * D_MODEL + lane * 16);
    float* ap = att + (size_t)d_ * D_MODEL + lane * 16;
    #pragma unroll
    for (int t = 0; t < 4; ++t) {
        f32x4 vv = vp[t];
        atomicAdd(ap + 4 * t + 0, sc * vv.x);
        atomicAdd(ap + 4 * t + 1, sc * vv.y);
        atomicAdd(ap + 4 * t + 2, sc * vv.z);
        atomicAdd(ap + 4 * t + 3, sc * vv.w);
    }
    if ((lane & 3) == 0) atomicAdd(z + (size_t)d_ * N_HEADS + (lane >> 2), sc);
}

// out_bf16[m,d] = att[m,d] / z[m, d/64]
__global__ __launch_bounds__(256) void attnorm_kernel(const float* __restrict__ att,
                                                      const float* __restrict__ z,
                                                      bf16_t* __restrict__ out, int total) {
    int i = blockIdx.x * 256 + threadIdx.x;
    if (i >= total) return;
    int m = i >> 9;               // / 512
    int h = (i & 511) >> 6;       // head
    out[i] = (bf16_t)(att[i] / z[m * N_HEADS + h]);
}

// ---------------- launcher ----------------
extern "C" void kernel_launch(void* const* d_in, const int* in_sizes, int n_in,
                              void* d_out, int out_size, void* d_ws, size_t ws_size,
                              hipStream_t stream) {
    (void)n_in; (void)out_size; (void)ws_size;
    const float* x    = (const float*)d_in[0];
    const float* mem  = (const float*)d_in[1];
    const float* ln0g = (const float*)d_in[2];
    const float* ln0b = (const float*)d_in[3];
    const float* ln1g = (const float*)d_in[4];
    const float* ln1b = (const float*)d_in[5];
    const float* ln2g = (const float*)d_in[6];
    const float* ln2b = (const float*)d_in[7];
    const float* Wq   = (const float*)d_in[8];
    const float* bq   = (const float*)d_in[9];
    const float* Wk   = (const float*)d_in[10];
    const float* bk   = (const float*)d_in[11];
    const float* Wv   = (const float*)d_in[12];
    const float* bv   = (const float*)d_in[13];
    const float* Wo   = (const float*)d_in[14];
    const float* bo   = (const float*)d_in[15];
    const float* W1   = (const float*)d_in[16];
    const float* b1   = (const float*)d_in[17];
    const float* W2   = (const float*)d_in[18];
    const float* b2   = (const float*)d_in[19];
    const int* src_self  = (const int*)d_in[20];
    const int* dst_self  = (const int*)d_in[21];
    const int* src_cross = (const int*)d_in[22];
    const int* dst_cross = (const int*)d_in[23];
    const int E_self  = in_sizes[20];
    const int E_cross = in_sizes[22];

    const int N = N_NODES, M = M_NODES, D = D_MODEL, DFF = D_FF, H = N_HEADS;

    // ---- carve workspace ----
    char* wsp = (char*)d_ws;
    auto carve = [&](size_t bytes) -> void* {
        void* p = (void*)wsp;
        wsp += (bytes + 255) & ~(size_t)255;
        return p;
    };
    float*  xcur  = (float*) carve((size_t)N * D * 4);
    bf16_t* hbf   = (bf16_t*)carve((size_t)N * D * 2);   // LN output / normalized att (reused)
    bf16_t* membf = (bf16_t*)carve((size_t)M * D * 2);
    float*  qf    = (float*) carve((size_t)N * D * 4);
    float*  kf    = (float*) carve((size_t)M * D * 4);
    float*  vf    = (float*) carve((size_t)M * D * 4);
    float*  att   = (float*) carve((size_t)N * D * 4);
    float*  zf    = (float*) carve((size_t)N * H * 4);
    bf16_t* WqT   = (bf16_t*)carve((size_t)D * D * 2);
    bf16_t* WkT   = (bf16_t*)carve((size_t)D * D * 2);
    bf16_t* WvT   = (bf16_t*)carve((size_t)D * D * 2);
    bf16_t* WoT   = (bf16_t*)carve((size_t)D * D * 2);
    bf16_t* W1T   = (bf16_t*)carve((size_t)DFF * D * 2);
    bf16_t* W2T   = (bf16_t*)carve((size_t)D * DFF * 2);
    bf16_t* ffh   = (bf16_t*)carve((size_t)N * DFF * 2);

    dim3 blk(256);
    auto g1 = [](int total) { return dim3((unsigned)((total + 255) / 256)); };

    // ---- one-time conversions ----
    transpose_bf16_kernel<<<g1(D * D), blk, 0, stream>>>(Wq, WqT, D, D, D * D);
    transpose_bf16_kernel<<<g1(D * D), blk, 0, stream>>>(Wk, WkT, D, D, D * D);
    transpose_bf16_kernel<<<g1(D * D), blk, 0, stream>>>(Wv, WvT, D, D, D * D);
    transpose_bf16_kernel<<<g1(D * D), blk, 0, stream>>>(Wo, WoT, D, D, D * D);
    transpose_bf16_kernel<<<g1(D * DFF), blk, 0, stream>>>(W1, W1T, D, DFF, D * DFF);
    transpose_bf16_kernel<<<g1(DFF * D), blk, 0, stream>>>(W2, W2T, DFF, D, DFF * D);
    f2bf_kernel<<<g1(M * D), blk, 0, stream>>>(mem, membf, M * D);

    dim3 gNxD(D / 128, N / 128);      // GEMM grid: N rows x D cols
    dim3 gMxD(D / 128, M / 128);
    dim3 gNxF(DFF / 128, N / 128);

    // ================= sublayer 0: self-attention =================
    ln_bf16_kernel<<<dim3(N), blk, 0, stream>>>(x, ln0g, ln0b, hbf, D);
    gemm_bf16_kernel<<<gNxD, blk, 0, stream>>>(hbf, WqT, bq, nullptr, qf, nullptr, D, D, 0);
    gemm_bf16_kernel<<<gNxD, blk, 0, stream>>>(hbf, WkT, bk, nullptr, kf, nullptr, D, D, 0);
    gemm_bf16_kernel<<<gNxD, blk, 0, stream>>>(hbf, WvT, bv, nullptr, vf, nullptr, D, D, 0);
    fill_zero_kernel<<<g1(N * D), blk, 0, stream>>>(att, N * D);
    fill_zero_kernel<<<g1(N * H), blk, 0, stream>>>(zf, N * H);
    edge_attn_kernel<<<dim3((E_self + 7) / 8), blk, 0, stream>>>(qf, kf, vf, src_self, dst_self, att, zf, E_self);
    attnorm_kernel<<<g1(N * D), blk, 0, stream>>>(att, zf, hbf, N * D);
    // xcur = x + (att/z) @ Wo + bo
    gemm_bf16_kernel<<<gNxD, blk, 0, stream>>>(hbf, WoT, bo, x, xcur, nullptr, D, D, 0);

    // ================= sublayer 1: cross-attention =================
    ln_bf16_kernel<<<dim3(N), blk, 0, stream>>>(xcur, ln1g, ln1b, hbf, D);
    gemm_bf16_kernel<<<gNxD, blk, 0, stream>>>(hbf, WqT, bq, nullptr, qf, nullptr, D, D, 0);
    gemm_bf16_kernel<<<gMxD, blk, 0, stream>>>(membf, WkT, bk, nullptr, kf, nullptr, D, D, 0);
    gemm_bf16_kernel<<<gMxD, blk, 0, stream>>>(membf, WvT, bv, nullptr, vf, nullptr, D, D, 0);
    fill_zero_kernel<<<g1(N * D), blk, 0, stream>>>(att, N * D);
    fill_zero_kernel<<<g1(N * H), blk, 0, stream>>>(zf, N * H);
    edge_attn_kernel<<<dim3((E_cross + 7) / 8), blk, 0, stream>>>(qf, kf, vf, src_cross, dst_cross, att, zf, E_cross);
    attnorm_kernel<<<g1(N * D), blk, 0, stream>>>(att, zf, hbf, N * D);
    // xcur = xcur + (att/z) @ Wo + bo   (in-place residual, each element touched once)
    gemm_bf16_kernel<<<gNxD, blk, 0, stream>>>(hbf, WoT, bo, xcur, xcur, nullptr, D, D, 0);

    // ================= sublayer 2: feed-forward =================
    ln_bf16_kernel<<<dim3(N), blk, 0, stream>>>(xcur, ln2g, ln2b, hbf, D);
    gemm_bf16_kernel<<<gNxF, blk, 0, stream>>>(hbf, W1T, b1, nullptr, nullptr, ffh, D, DFF, 1);
    gemm_bf16_kernel<<<gNxD, blk, 0, stream>>>(ffh, W2T, b2, xcur, (float*)d_out, nullptr, DFF, D, 0);
}
